// Network_44152263803532
// MI455X (gfx1250) — matrix operand order repaired and checked by reading, not compiled, not run on hardware
//
#include <hip/hip_runtime.h>

typedef __attribute__((ext_vector_type(16))) __bf16       v16bf;
typedef __attribute__((ext_vector_type(8)))  float        v8f;
typedef __attribute__((ext_vector_type(4)))  __bf16       v4bf;
typedef __attribute__((ext_vector_type(4)))  unsigned int v4u;
typedef __attribute__((ext_vector_type(4)))  float        v4f;

union FragBF { v16bf v; v4u u[2]; };

#define NNODE 8192
#define CCH   128
#define MROW  256

__device__ __forceinline__ float sigmoidf_(float x) { return 1.0f / (1.0f + __expf(-x)); }

__device__ __forceinline__ FragBF ldFragA(const __bf16* p) {   // 16 contiguous K values (per lane-half base)
    FragBF f; f.u[0] = *(const v4u*)(p); f.u[1] = *(const v4u*)(p + 8); return f;
}
__device__ __forceinline__ FragBF ldFragB(const __bf16* p) {   // 16 contiguous K values
    FragBF f; f.u[0] = *(const v4u*)(p); f.u[1] = *(const v4u*)(p + 8); return f;
}

// ---------------- f32 -> bf16 convert (4 elems / thread) ----------------
__global__ void cvt_bf16_x4(const float* __restrict__ in, __bf16* __restrict__ out, long n4)
{
    long t = (long)blockIdx.x * blockDim.x + threadIdx.x;
    if (t >= n4) return;
    v4f f = ((const v4f*)in)[t];
    v4bf o = { (__bf16)f.x, (__bf16)f.y, (__bf16)f.z, (__bf16)f.w };
    ((v4bf*)out)[t] = o;
}

// ---------------- embedding gather: h, h_bf16, hT_bf16 ----------------
__global__ void embed_kernel(const int* __restrict__ atoms, const float* __restrict__ emb,
                             float* __restrict__ h, __bf16* __restrict__ hbf,
                             __bf16* __restrict__ hT)
{
    long t = (long)blockIdx.x * blockDim.x + threadIdx.x;   // N*C
    int n = (int)(t >> 7), c = (int)(t & 127);
    float v = emb[(long)atoms[n] * CCH + c];
    h[t] = v;
    hbf[t] = (__bf16)v;
    hT[(long)c * NNODE + n] = (__bf16)v;
}

// ---------------- GRU elementwise update ----------------
__global__ void gru_kernel(const float* __restrict__ gi, const float* __restrict__ gh,
                           const float* __restrict__ hin, float* __restrict__ hout,
                           __bf16* __restrict__ hbf, __bf16* __restrict__ hT)
{
    long t = (long)blockIdx.x * blockDim.x + threadIdx.x;   // N*C
    int n = (int)(t >> 7), c = (int)(t & 127);
    long gb = (long)n * 384 + c;
    float ir = gi[gb], iz = gi[gb + 128], in_ = gi[gb + 256];
    float hr = gh[gb], hz = gh[gb + 128], hn = gh[gb + 256];
    float h  = hin[t];
    float r  = sigmoidf_(ir + hr);
    float z  = sigmoidf_(iz + hz);
    float nn = tanhf(in_ + r * hn);
    float hv = (1.0f - z) * nn + z * h;
    hout[t] = hv;
    hbf[t] = (__bf16)hv;
    hT[(long)c * NNODE + n] = (__bf16)hv;
}

// ---------------- generic WMMA GEMM: out[r,j] = sum_k A[r,k]*Bw[j,k] + bias[j] ----------------
// A bf16 row-major [R,K]; Bw bf16 row-major [J,K]; grid = (R/256, J/64); block = 256 (8 waves).
// Each wave: 2 row-tiles x 4 col-tiles (32 rows x 64 cols). Double-buffered k-loop with
// pointer-bump addressing: all loads use immediate offsets 0/+16B/+64B/+80B off 6 bumped
// per-lane pointers. Requires K % 64 == 0. NOTE: the final iteration's look-ahead loads read
// up to 96B past A/Bw; all GEMM operands live inside d_ws with later allocations behind them.
__global__ void __launch_bounds__(256)
gemm_nt_wmma(const __bf16* __restrict__ A, const __bf16* __restrict__ Bw,
             const float* __restrict__ bias,
             float* __restrict__ outF, __bf16* __restrict__ outH,
             int K, int ldo)
{
    const int lane = threadIdx.x & 31;
    const int wave = threadIdx.x >> 5;
    const int lm = lane & 15;
    const int hi = lane >> 4;                       // 0: lanes 0-15, 1: lanes 16-31
    const long r0 = (long)blockIdx.x * 256 + wave * 32;
    const int  c0 = blockIdx.y * 64;

    v8f acc[2][4];
    v8f zero = { 0.f, 0.f, 0.f, 0.f, 0.f, 0.f, 0.f, 0.f };
#pragma unroll
    for (int r = 0; r < 2; ++r)
#pragma unroll
        for (int t = 0; t < 4; ++t) acc[r][t] = zero;

    const __bf16* pa0 = A + (r0 + lm) * (long)K + hi * 16;
    const __bf16* pa1 = pa0 + 16 * (long)K;
    const __bf16* pb0 = Bw + (long)(c0 +  0 + lm) * K + hi * 16;
    const __bf16* pb1 = Bw + (long)(c0 + 16 + lm) * K + hi * 16;
    const __bf16* pb2 = Bw + (long)(c0 + 32 + lm) * K + hi * 16;
    const __bf16* pb3 = Bw + (long)(c0 + 48 + lm) * K + hi * 16;

    FragBF aA[2], bA[4], aB[2], bB[4];
    aA[0] = ldFragA(pa0);  aA[1] = ldFragA(pa1);
    bA[0] = ldFragB(pb0);  bA[1] = ldFragB(pb1);
    bA[2] = ldFragB(pb2);  bA[3] = ldFragB(pb3);

    const int iters = K >> 6;
    for (int it = 0; it < iters; ++it) {
        // stage-B loads (k+32) at fixed +64B/+80B immediates
        aB[0] = ldFragA(pa0 + 32);  aB[1] = ldFragA(pa1 + 32);
        bB[0] = ldFragB(pb0 + 32);  bB[1] = ldFragB(pb1 + 32);
        bB[2] = ldFragB(pb2 + 32);  bB[3] = ldFragB(pb3 + 32);
        __builtin_prefetch(pa0 + 1024, 0, 1);       // stream-ahead for the big streaming A
#pragma unroll
        for (int r = 0; r < 2; ++r)
#pragma unroll
            for (int t = 0; t < 4; ++t)
                acc[r][t] = __builtin_amdgcn_wmma_f32_16x16x32_bf16(
                    false, aA[r].v, false, bA[t].v, (short)0, acc[r][t], false, false);

        pa0 += 64; pa1 += 64;
        pb0 += 64; pb1 += 64; pb2 += 64; pb3 += 64;

        // stage-A loads for next k (dead on last iteration; in-bounds of d_ws)
        aA[0] = ldFragA(pa0);  aA[1] = ldFragA(pa1);
        bA[0] = ldFragB(pb0);  bA[1] = ldFragB(pb1);
        bA[2] = ldFragB(pb2);  bA[3] = ldFragB(pb3);
#pragma unroll
        for (int r = 0; r < 2; ++r)
#pragma unroll
            for (int t = 0; t < 4; ++t)
                acc[r][t] = __builtin_amdgcn_wmma_f32_16x16x32_bf16(
                    false, aB[r].v, false, bB[t].v, (short)0, acc[r][t], false, false);
    }

#pragma unroll
    for (int t = 0; t < 4; ++t) {
        int col = c0 + t * 16 + lm;
        float bv = bias ? bias[col] : 0.0f;
#pragma unroll
        for (int r = 0; r < 2; ++r) {
#pragma unroll
            for (int g = 0; g < 8; ++g) {
                long row = r0 + r * 16 + g + hi * 8;
                float v = acc[r][t][g] + bv;
                if (outF) outF[row * (long)ldo + col] = v;
                if (outH) outH[row * (long)ldo + col] = (__bf16)v;
            }
        }
    }
}

// ---------------- membership/h_final -> padded NHWC conv input (torch .view shuffle) ----------------
__global__ void build_a0(const float* __restrict__ hf, __bf16* __restrict__ a0)
{
    long t = (long)blockIdx.x * blockDim.x + threadIdx.x;   // 128*256*256, c fastest
    int c = (int)(t & 127);
    int j = (int)((t >> 7) & 255);
    int i = (int)(t >> 15);
    long f = (long)c * 65536 + i * 256 + j;                  // NCHW flat index
    int is = (int)(f >> 15), js = (int)((f >> 7) & 255), cs = (int)(f & 127);
    float v = 0.5f * (hf[is * CCH + cs] + hf[js * CCH + cs]);
    a0[((long)(i + 1) * 258 + (j + 1)) * CCH + c] = (__bf16)v;
}

// ---------------- conv weight prepack: Wp[ky][kx][co_pad][ci_pad] bf16, zero-padded ----------------
__global__ void pack_conv_w(const float* __restrict__ W, __bf16* __restrict__ Wp,
                            int co, int ci, int coPad, int ciPad)
{
    int t = blockIdx.x * blockDim.x + threadIdx.x;
    int total = 9 * coPad * ciPad;
    if (t >= total) return;
    int i = t % ciPad;
    int o = (t / ciPad) % coPad;
    int kk = t / (ciPad * coPad);
    int ky = kk / 3, kx = kk % 3;
    float v = (o < co && i < ci) ? W[(((long)o * ci + i) * 3 + ky) * 3 + kx] : 0.0f;
    Wp[t] = (__bf16)v;
}

// ---------------- implicit-GEMM WMMA conv 3x3 SAME over NHWC padded (258x258xCPIN) ----------------
// grid = (16, 256); block = 32 * (coPad/16); wave w -> co tile w. CPIN compile-time for full unroll.
template<int CPIN>
__global__ void __launch_bounds__(128)
conv3x3_wmma(const __bf16* __restrict__ act,
             const __bf16* __restrict__ Wp, const float* __restrict__ bias,
             int coReal, int coPad,
             __bf16* __restrict__ outAct, int cpOut,
             float* __restrict__ outNCHW, int doRelu)
{
    const int lane = threadIdx.x & 31;
    const int wave = threadIdx.x >> 5;
    const int lm = lane & 15;
    const int hi = lane >> 4;
    const int y  = blockIdx.y;
    const int x0 = blockIdx.x * 16;
    const int co0 = wave * 16;

    v8f acc = { 0.f, 0.f, 0.f, 0.f, 0.f, 0.f, 0.f, 0.f };

#pragma unroll
    for (int ky = 0; ky < 3; ++ky) {
#pragma unroll
        for (int kx = 0; kx < 3; ++kx) {
            const __bf16* aBase = act + ((long)(y + ky) * 258 + (x0 + kx + lm)) * CPIN + hi * 16;
            const __bf16* wBase = Wp + ((long)(ky * 3 + kx) * coPad + co0 + lm) * CPIN + hi * 16;
#pragma unroll
            for (int ci = 0; ci < CPIN; ci += 32) {
                FragBF a = ldFragA(aBase + ci);
                FragBF b = ldFragB(wBase + ci);
                acc = __builtin_amdgcn_wmma_f32_16x16x32_bf16(
                    false, a.v, false, b.v, (short)0, acc, false, false);
            }
        }
    }

    int co = co0 + lm;
    if (co < coReal) {                 // divergence only after all WMMA (EXEC all-1s during wmma)
        float bv = bias[co];
#pragma unroll
        for (int g = 0; g < 8; ++g) {
            int px = x0 + g + hi * 8;
            float v = acc[g] + bv;
            if (doRelu) v = fmaxf(v, 0.0f);
            if (outAct)  outAct[((long)(y + 1) * 258 + (px + 1)) * cpOut + co] = (__bf16)v;
            if (outNCHW) outNCHW[(long)co * 65536 + y * 256 + px] = v;
        }
    }
}

// ---------------- final symmetrize + sigmoid ----------------
__global__ void sym_sigmoid(const float* __restrict__ o6, float* __restrict__ out)
{
    int t = blockIdx.x * blockDim.x + threadIdx.x;   // 12*256*256
    int c = t >> 16, i = (t >> 8) & 255, j = t & 255;
    float a = o6[t];
    float b = o6[(c << 16) + (j << 8) + i];
    out[t] = sigmoidf_(0.5f * (a + b));
}

// =======================================================================
extern "C" void kernel_launch(void* const* d_in, const int* in_sizes, int n_in,
                              void* d_out, int out_size, void* d_ws, size_t ws_size,
                              hipStream_t stream)
{
    const float* adj   = (const float*)d_in[0];
    const int*   atoms = (const int*)  d_in[1];
    const float* mem   = (const float*)d_in[2];
    const float* emb   = (const float*)d_in[3];
    const float* Wl    = (const float*)d_in[4];
    const float* bl    = (const float*)d_in[5];
    const float* Wih   = (const float*)d_in[6];
    const float* bih   = (const float*)d_in[7];
    const float* Whh   = (const float*)d_in[8];
    const float* bhh   = (const float*)d_in[9];
    const float* Wc[6]; const float* bc[6];
    for (int i = 0; i < 6; ++i) { Wc[i] = (const float*)d_in[10 + 2 * i]; bc[i] = (const float*)d_in[11 + 2 * i]; }

    char* p = (char*)d_ws;
    auto alloc = [&](size_t bytes) -> char* {
        char* r = p; p += (bytes + 255) & ~(size_t)255; return r;
    };

    __bf16* adjb  = (__bf16*)alloc((size_t)NNODE * NNODE * 2);      // 128 MB, L2-resident bf16 adjacency
    __bf16* memb  = (__bf16*)alloc((size_t)MROW * NNODE * 2);
    __bf16* wlb   = (__bf16*)alloc((size_t)CCH * CCH * 2);
    __bf16* wihb  = (__bf16*)alloc((size_t)3 * CCH * CCH * 2);
    __bf16* whhb  = (__bf16*)alloc((size_t)3 * CCH * CCH * 2);
    float*  hA    = (float*) alloc((size_t)NNODE * CCH * 4);
    float*  hB    = (float*) alloc((size_t)NNODE * CCH * 4);
    __bf16* hbf   = (__bf16*)alloc((size_t)NNODE * CCH * 2);
    __bf16* hT    = (__bf16*)alloc((size_t)NNODE * CCH * 2);
    __bf16* ybf   = (__bf16*)alloc((size_t)NNODE * CCH * 2);
    __bf16* xbf   = (__bf16*)alloc((size_t)NNODE * CCH * 2);
    float*  gi    = (float*) alloc((size_t)NNODE * 384 * 4);
    float*  gh    = (float*) alloc((size_t)NNODE * 384 * 4);
    float*  hfin  = (float*) alloc((size_t)MROW * CCH * 4);
    __bf16* wp1   = (__bf16*)alloc((size_t)9 * 64 * 128 * 2);
    __bf16* wp2   = (__bf16*)alloc((size_t)9 * 32 * 64 * 2);
    __bf16* wp3   = (__bf16*)alloc((size_t)9 * 32 * 32 * 2);
    __bf16* wp4   = (__bf16*)alloc((size_t)9 * 32 * 32 * 2);
    __bf16* wp5   = (__bf16*)alloc((size_t)9 * 32 * 32 * 2);
    __bf16* wp6   = (__bf16*)alloc((size_t)9 * 16 * 32 * 2);
    size_t a0B = (size_t)258 * 258 * 128 * 2;
    size_t a1B = (size_t)258 * 258 * 64 * 2;
    size_t a2B = (size_t)258 * 258 * 32 * 2;
    __bf16* a0 = (__bf16*)alloc(a0B);
    __bf16* a1 = (__bf16*)alloc(a1B);
    __bf16* a2 = (__bf16*)alloc(a2B);
    __bf16* a3 = (__bf16*)alloc(a2B);
    __bf16* a4 = (__bf16*)alloc(a2B);
    __bf16* a5 = (__bf16*)alloc(a2B);
    float*  o6 = (float*) alloc((size_t)12 * 65536 * 4);

    if ((size_t)(p - (char*)d_ws) > ws_size) return;  // scratch too small: do nothing (deterministic)

    // Zero padded activation halos / pad channels (captured as memset nodes).
    hipMemsetAsync(a0, 0, a0B, stream);
    hipMemsetAsync(a1, 0, a1B, stream);
    hipMemsetAsync(a2, 0, a2B, stream);
    hipMemsetAsync(a3, 0, a2B, stream);
    hipMemsetAsync(a4, 0, a2B, stream);
    hipMemsetAsync(a5, 0, a2B, stream);

    // ---- precision conversion / packing ----
    {
        long n4 = (long)NNODE * NNODE / 4;
        cvt_bf16_x4<<<(unsigned)((n4 + 255) / 256), 256, 0, stream>>>(adj, adjb, n4);
        n4 = (long)MROW * NNODE / 4;
        cvt_bf16_x4<<<(unsigned)((n4 + 255) / 256), 256, 0, stream>>>(mem, memb, n4);
        n4 = (long)CCH * CCH / 4;
        cvt_bf16_x4<<<(unsigned)((n4 + 255) / 256), 256, 0, stream>>>(Wl, wlb, n4);
        n4 = (long)3 * CCH * CCH / 4;
        cvt_bf16_x4<<<(unsigned)((n4 + 255) / 256), 256, 0, stream>>>(Wih, wihb, n4);
        cvt_bf16_x4<<<(unsigned)((n4 + 255) / 256), 256, 0, stream>>>(Whh, whhb, n4);
    }
    pack_conv_w<<<(9 * 64 * 128 + 255) / 256, 256, 0, stream>>>(Wc[0], wp1, 50, 128, 64, 128);
    pack_conv_w<<<(9 * 32 * 64  + 255) / 256, 256, 0, stream>>>(Wc[1], wp2, 25, 50, 32, 64);
    pack_conv_w<<<(9 * 32 * 32  + 255) / 256, 256, 0, stream>>>(Wc[2], wp3, 25, 25, 32, 32);
    pack_conv_w<<<(9 * 32 * 32  + 255) / 256, 256, 0, stream>>>(Wc[3], wp4, 25, 25, 32, 32);
    pack_conv_w<<<(9 * 32 * 32  + 255) / 256, 256, 0, stream>>>(Wc[4], wp5, 25, 25, 32, 32);
    pack_conv_w<<<(9 * 16 * 32  + 255) / 256, 256, 0, stream>>>(Wc[5], wp6, 12, 25, 16, 32);

    // ---- embedding gather ----
    const int NC = NNODE * CCH;
    embed_kernel<<<NC / 256, 256, 0, stream>>>(atoms, emb, hA, hbf, hT);

    // ---- 4 graph-propagation steps ----
    float* hin = hA; float* hout = hB;
    for (int step = 0; step < 4; ++step) {
        // y = adjacency @ h        (A = adj_bf16 [8192,8192], Bw = hT [128,8192])
        gemm_nt_wmma<<<dim3(NNODE / 256, CCH / 64), 256, 0, stream>>>(
            adjb, hT, nullptr, nullptr, ybf, NNODE, CCH);
        // x_in = y @ Wl^T + bl     (Bw = Wl [128,128])
        gemm_nt_wmma<<<dim3(NNODE / 256, CCH / 64), 256, 0, stream>>>(
            ybf, wlb, bl, nullptr, xbf, CCH, CCH);
        // gi = x_in @ Wih^T + bih  (Bw = Wih [384,128])
        gemm_nt_wmma<<<dim3(NNODE / 256, 384 / 64), 256, 0, stream>>>(
            xbf, wihb, bih, gi, nullptr, CCH, 384);
        // gh = h @ Whh^T + bhh
        gemm_nt_wmma<<<dim3(NNODE / 256, 384 / 64), 256, 0, stream>>>(
            hbf, whhb, bhh, gh, nullptr, CCH, 384);
        // GRU update -> new h (fp32 + bf16 + transposed bf16)
        gru_kernel<<<NC / 256, 256, 0, stream>>>(gi, gh, hin, hout, hbf, hT);
        float* t = hin; hin = hout; hout = t;
    }

    // ---- h_final = membership @ h  (A = memb [256,8192], Bw = hT [128,8192]) ----
    gemm_nt_wmma<<<dim3(MROW / 256, CCH / 64), 256, 0, stream>>>(
        memb, hT, nullptr, hfin, nullptr, NNODE, CCH);

    // ---- conv input via the torch .view channel shuffle, into padded NHWC ----
    build_a0<<<(128 * 65536) / 256, 256, 0, stream>>>(hfin, a0);

    // ---- six 3x3 convs ----
    conv3x3_wmma<128><<<dim3(16, 256), 128, 0, stream>>>(a0, wp1, bc[0], 50, 64, a1, 64, nullptr, 1);
    conv3x3_wmma< 64><<<dim3(16, 256),  64, 0, stream>>>(a1, wp2, bc[1], 25, 32, a2, 32, nullptr, 1);
    conv3x3_wmma< 32><<<dim3(16, 256),  64, 0, stream>>>(a2, wp3, bc[2], 25, 32, a3, 32, nullptr, 1);
    conv3x3_wmma< 32><<<dim3(16, 256),  64, 0, stream>>>(a3, wp4, bc[3], 25, 32, a4, 32, nullptr, 1);
    conv3x3_wmma< 32><<<dim3(16, 256),  64, 0, stream>>>(a4, wp5, bc[4], 25, 32, a5, 32, nullptr, 1);
    conv3x3_wmma< 32><<<dim3(16, 256),  32, 0, stream>>>(a5, wp6, bc[5], 12, 16, nullptr, 0, o6, 0);

    // ---- symmetrize + sigmoid -> d_out [12,256,256] fp32 ----
    sym_sigmoid<<<(12 * 65536) / 256, 256, 0, stream>>>(o6, (float*)d_out);
}
